// Resonator_7121055776898
// MI455X (gfx1250) — 8-bit matrix operand order repaired and checked by reading, compiled, not run on hardware
//
#include <hip/hip_runtime.h>
#include <hip/hip_bf16.h>
#include <stdint.h>


#define F_ 3
#define V_ 512
#define D_ 8192
#define B_ 64
#define ITERS_ 10
#define NT2 4

typedef __attribute__((ext_vector_type(8))) int v8i;

static __device__ __forceinline__ v8i load_fragA(const int8_t* base, int row0,
                                                 int ld, int k0) {
  const int lane = threadIdx.x & 31;
  const int r    = lane & 15;
  const int half = lane >> 4;
  const int8_t* p = base + (size_t)(row0 + r) * (size_t)ld + k0 + half * 8;
  v8i out;
#pragma unroll
  for (int j = 0; j < 8; ++j) {
    out[j] = *(const int*)(p + (j >> 1) * 16 + (j & 1) * 4);
  }
  return out;
}

static __device__ __forceinline__ v8i load_fragB(const int8_t* base, int col0,
                                                 int ld, int k0) {
  const int lane = threadIdx.x & 31;
  const int c    = lane & 15;
  const int half = lane >> 4;
  const int8_t* p = base + (size_t)(col0 + c) * (size_t)ld + k0 + half * 8;
  v8i out;
#pragma unroll
  for (int j = 0; j < 8; ++j) {
    out[j] = *(const int*)(p + (j >> 1) * 16 + (j & 1) * 4);
  }
  return out;
}

__global__ void pack_cb_kernel(const float* __restrict__ cb,
                               int8_t* __restrict__ cb8,
                               int8_t* __restrict__ cbT8) {
  size_t i = (size_t)blockIdx.x * blockDim.x + threadIdx.x;
  const size_t total = (size_t)F_ * V_ * D_;
  if (i >= total) return;
  float x = cb[i];
  int8_t s = (int8_t)((x > 0.f) - (x < 0.f));
  cb8[i] = s;
  int    f   = (int)(i / ((size_t)V_ * D_));
  size_t rem = i - (size_t)f * ((size_t)V_ * D_);
  int v = (int)(rem / D_);
  int d = (int)(rem % D_);
  cbT8[(size_t)f * D_ * V_ + (size_t)d * V_ + v] = s;
}

__global__ void pack_in_kernel(const float* __restrict__ inp,
                               int8_t* __restrict__ in8) {
  size_t i = (size_t)blockIdx.x * blockDim.x + threadIdx.x;
  if (i >= (size_t)B_ * D_) return;
  float x = inp[i];
  in8[i] = (int8_t)((x > 0.f) - (x < 0.f));
}

__global__ void pack_est_kernel(const float* __restrict__ ie,
                                int8_t* __restrict__ est) {
  size_t i = (size_t)blockIdx.x * blockDim.x + threadIdx.x;
  const size_t total = (size_t)F_ * B_ * D_;
  if (i >= total) return;
  int    f   = (int)(i / ((size_t)B_ * D_));
  size_t rem = i - (size_t)f * ((size_t)B_ * D_);
  int b = (int)(rem / D_);
  int d = (int)(rem % D_);
  float x = ie[((size_t)b * F_ + f) * D_ + d];
  est[i] = (int8_t)((x > 0.f) - (x < 0.f));
}

__global__ void zero_flags_kernel(int* flags) {
  if (threadIdx.x < ITERS_) flags[threadIdx.x] = 0;
}

__global__ void ne_kernel(const int8_t* __restrict__ in8,
                          const int8_t* __restrict__ est,
                          int8_t* __restrict__ ne) {
  size_t i = (size_t)blockIdx.x * blockDim.x + threadIdx.x;
  const size_t total = (size_t)F_ * B_ * D_ / 4;
  if (i >= total) return;
  int    f   = (int)(i / ((size_t)B_ * D_ / 4));
  size_t rem = i - (size_t)f * ((size_t)B_ * D_ / 4);
  int f1 = (f + 1) % 3, f2 = (f + 2) % 3;
  int a  = *(const int*)(in8 + rem * 4);
  int e1 = *(const int*)(est + (size_t)f1 * B_ * D_ + rem * 4);
  int e2 = *(const int*)(est + (size_t)f2 * B_ * D_ + rem * 4);
  int out;
  const int8_t* pa = (const int8_t*)&a;
  const int8_t* p1 = (const int8_t*)&e1;
  const int8_t* p2 = (const int8_t*)&e2;
  int8_t* po = (int8_t*)&out;
#pragma unroll
  for (int j = 0; j < 4; ++j) po[j] = (int8_t)(pa[j] * p1[j] * p2[j]);
  *(int*)(ne + i * 4) = out;
}

__global__ __launch_bounds__(32) void gemm1_kernel(
    const int8_t* __restrict__ A  ,
    const int8_t* __restrict__ CB  ,
    int8_t* __restrict__ simLo, int8_t* __restrict__ simHi,
    int* __restrict__ sim32, int write32) {
  const int f  = blockIdx.z;
  const int n0 = blockIdx.x * 16;
  const int m0 = blockIdx.y * 16;
  const int8_t* Af = A  + (size_t)f * B_ * D_;
  const int8_t* Bf = CB + (size_t)f * V_ * D_;
  v8i acc = {};
#pragma unroll 4
  for (int k0 = 0; k0 < D_; k0 += 64) {
    v8i a = load_fragA(Af, m0, D_, k0);
    v8i b = load_fragB(Bf, n0, D_, k0);
    acc = __builtin_amdgcn_wmma_i32_16x16x64_iu8(true, a, true, b, acc,
                                                 false, false);
  }
  const int lane = threadIdx.x & 31;
  const int half = lane >> 4;
  const int n    = n0 + (lane & 15);
#pragma unroll
  for (int j = 0; j < 8; ++j) {
    int m = m0 + j + half * 8;
    int s = acc[j];
    int lo = ((s + 128) & 255) - 128;
    int hi = (s - lo) >> 8;
    size_t idx = ((size_t)f * B_ + m) * V_ + n;
    simLo[idx] = (int8_t)lo;
    simHi[idx] = (int8_t)hi;
    if (write32) sim32[idx] = s;
  }
}

__global__ __launch_bounds__(32) void gemm2_kernel(
    const int8_t* __restrict__ simLo, const int8_t* __restrict__ simHi,
    const int8_t* __restrict__ CBT  ,
    int8_t* __restrict__ est  , int* __restrict__ changedFlag) {
  const int f  = blockIdx.z;
  const int n0 = blockIdx.x * (16 * NT2);
  const int m0 = blockIdx.y * 16;
  const int8_t* ALo = simLo + (size_t)f * B_ * V_;
  const int8_t* AHi = simHi + (size_t)f * B_ * V_;
  const int8_t* Bf  = CBT   + (size_t)f * D_ * V_;
  v8i accLo[NT2] = {};
  v8i accHi[NT2] = {};
#pragma unroll
  for (int k0 = 0; k0 < V_; k0 += 64) {
    v8i alo = load_fragA(ALo, m0, V_, k0);
    v8i ahi = load_fragA(AHi, m0, V_, k0);
#pragma unroll
    for (int t = 0; t < NT2; ++t) {
      v8i b = load_fragB(Bf, n0 + t * 16, V_, k0);
      accLo[t] = __builtin_amdgcn_wmma_i32_16x16x64_iu8(true, alo, true, b,
                                                        accLo[t], false, false);
      accHi[t] = __builtin_amdgcn_wmma_i32_16x16x64_iu8(true, ahi, true, b,
                                                        accHi[t], false, false);
    }
  }
  const int lane = threadIdx.x & 31;
  const int half = lane >> 4;
  int anyChanged = 0;
#pragma unroll
  for (int t = 0; t < NT2; ++t) {
    const int n = n0 + t * 16 + (lane & 15);
#pragma unroll
    for (int j = 0; j < 8; ++j) {
      int m = m0 + j + half * 8;
      int val = accLo[t][j] + (accHi[t][j] << 8);
      int8_t s = (int8_t)((val > 0) - (val < 0));
      size_t idx = ((size_t)f * B_ + m) * D_ + n;
      int8_t old = est[idx];
      anyChanged |= (old != s);
      est[idx] = s;
    }
  }
  if (anyChanged) atomicOr(changedFlag, 1);
}

__global__ void argmax_kernel(const int* __restrict__ sim32,
                              float* __restrict__ out) {
  int t = blockIdx.x * blockDim.x + threadIdx.x;
  if (t >= B_ * F_) return;
  int b = t / F_, f = t % F_;
  const int* p = sim32 + ((size_t)f * B_ + b) * V_;
  int best = p[0], bi = 0;
  for (int v = 1; v < V_; ++v) {
    int s = p[v];
    if (s > best) { best = s; bi = v; }
  }
  out[t] = (float)bi;
}

__global__ void cnt_kernel(const int* __restrict__ flags,
                           float* __restrict__ out) {
  if (threadIdx.x == 0 && blockIdx.x == 0) {
    int cnt = ITERS_;
    for (int i = 0; i < ITERS_; ++i) {
      if (!flags[i]) { cnt = i + 1; break; }
    }
    out[B_ * F_] = (float)(cnt - 1);
  }
}

extern "C" void kernel_launch(void* const* d_in, const int* in_sizes, int n_in,
                              void* d_out, int out_size, void* d_ws,
                              size_t ws_size, hipStream_t stream) {
  const float* inputs    = (const float*)d_in[0];
  const float* init_est  = (const float*)d_in[1];
  const float* codebooks = (const float*)d_in[2];
  float* out = (float*)d_out;

  uint8_t* ws = (uint8_t*)d_ws;
  size_t off = 0;
  auto carve = [&](size_t bytes) -> void* {
    void* p = ws + off;
    off += (bytes + 255) & ~(size_t)255;
    return p;
  };
  int8_t* cb8   = (int8_t*)carve((size_t)F_ * V_ * D_);
  int8_t* cbT8  = (int8_t*)carve((size_t)F_ * D_ * V_);
  int8_t* in8   = (int8_t*)carve((size_t)B_ * D_);
  int8_t* est   = (int8_t*)carve((size_t)F_ * B_ * D_);
  int8_t* ne    = (int8_t*)carve((size_t)F_ * B_ * D_);
  int8_t* simLo = (int8_t*)carve((size_t)F_ * B_ * V_);
  int8_t* simHi = (int8_t*)carve((size_t)F_ * B_ * V_);
  int*    sim32 = (int*)carve((size_t)F_ * B_ * V_ * sizeof(int));
  int*    flags = (int*)carve(ITERS_ * sizeof(int));
  (void)ws_size; (void)in_sizes; (void)n_in; (void)out_size;

  {
    size_t t = (size_t)F_ * V_ * D_;
    pack_cb_kernel<<<(unsigned)((t + 255) / 256), 256, 0, stream>>>(codebooks,
                                                                    cb8, cbT8);
  }
  {
    size_t t = (size_t)B_ * D_;
    pack_in_kernel<<<(unsigned)((t + 255) / 256), 256, 0, stream>>>(inputs,
                                                                    in8);
  }
  {
    size_t t = (size_t)F_ * B_ * D_;
    pack_est_kernel<<<(unsigned)((t + 255) / 256), 256, 0, stream>>>(init_est,
                                                                     est);
  }
  zero_flags_kernel<<<1, 32, 0, stream>>>(flags);

  const dim3 g1(V_ / 16, B_ / 16, F_);
  const dim3 g2(D_ / (16 * NT2), B_ / 16, F_);
  for (int it = 0; it < ITERS_; ++it) {
    size_t t = (size_t)F_ * B_ * D_ / 4;
    ne_kernel<<<(unsigned)((t + 255) / 256), 256, 0, stream>>>(in8, est, ne);
    gemm1_kernel<<<g1, 32, 0, stream>>>(ne, cb8, simLo, simHi, sim32, 0);
    gemm2_kernel<<<g2, 32, 0, stream>>>(simLo, simHi, cbT8, est, flags + it);
  }

  gemm1_kernel<<<g1, 32, 0, stream>>>(est, cb8, simLo, simHi, sim32, 1);
  argmax_kernel<<<1, B_ * F_, 0, stream>>>(sim32, out);
  cnt_kernel<<<1, 1, 0, stream>>>(flags, out);
}
